// LFQAutoEncoder_45148696216476
// MI455X (gfx1250) — matrix operand order repaired and checked by reading, not compiled, not run on hardware
//
#include <hip/hip_runtime.h>

typedef __attribute__((ext_vector_type(16))) _Float16 v16h;
typedef __attribute__((ext_vector_type(8)))  _Float16 v8h;
typedef __attribute__((ext_vector_type(8)))  float    v8f;
typedef __attribute__((ext_vector_type(4)))  int      v4i;

#if defined(__gfx1250__) && __has_builtin(__builtin_amdgcn_global_load_async_to_lds_b128)
#define ASYNC_LDS 1
typedef __attribute__((address_space(1))) v4i gv4i;   // global 16B chunk
typedef __attribute__((address_space(3))) v4i lv4i;   // LDS 16B chunk
#else
#define ASYNC_LDS 0
#endif

__device__ __forceinline__ float gelu_exact(float x) {
    return 0.5f * x * (1.0f + erff(x * 0.70710678118654752f));
}

__device__ __forceinline__ void wait_async_lds() {
#if ASYNC_LDS
#if __has_builtin(__builtin_amdgcn_s_wait_asynccnt)
    __builtin_amdgcn_s_wait_asynccnt(0);
#else
    asm volatile("s_wait_asynccnt 0x0" ::: "memory");
#endif
#endif
}

// ---------------------------------------------------------------------------
// Workspace layout (bytes):
//   [0      , 32768 ) packed f16 weights (im2col K-major per output channel)
//   [32768  , 36864 ) stats: pse[2], commit[2], avg_prob[2*256]  (516 floats)
//   [36864  , +16MB ) bufA: out1 f16 (16MB) -> later hn f16 (8MB) + h f32 (4MB) + q f16 (2MB)
//   [bufA+16MB, +32MB) bufB: conv2 f16 (32MB) -> later d1out f16 (8MB) + d2out f16 (16MB)
// ---------------------------------------------------------------------------
#define WP_C2 0
#define WP_C3 5120
#define WP_D1 5632
#define WP_D2 8704
#define WP_TOTAL 13312
#define NTOK_LFQ 131072           // 32*64*64
#define Y_ELEMS  2097152          // 32*1*256*256
#define IDX_OFF  2097152
#define AUX_POS  2359296          // IDX_OFF + 32*64*64*2

// ---------------------------------------------------------------------------
// Prep: pack conv weights to f16 [Coutpad][Kpad] (zero padded), zero stats.
// ---------------------------------------------------------------------------
__global__ __launch_bounds__(256)
void prep_pack(const float* __restrict__ w2, const float* __restrict__ w3,
               const float* __restrict__ d1w, const float* __restrict__ d2w,
               _Float16* __restrict__ wp, float* __restrict__ stats)
{
    int t = blockIdx.x * 256 + threadIdx.x;
    if (t < 5120) {                                  // conv2: Kreal=144, Kpad=160
        int m = t / 160, k = t % 160;
        float v = 0.f;
        if (k < 144) { int ci = k / 9, tap = k - ci * 9; v = w2[(m * 16 + ci) * 9 + tap]; }
        wp[WP_C2 + t] = (_Float16)v;
    } else if (t < 5632) {                           // conv3: 1x1, 8->pad16, K=32
        int i = t - 5120; int m = i / 32, k = i % 32;
        float v = (m < 8) ? w3[m * 32 + k] : 0.f;
        wp[WP_C3 + i] = (_Float16)v;
    } else if (t < 8704) {                           // d1: Kreal=72, Kpad=96
        int i = t - 5632; int m = i / 96, k = i % 96;
        float v = 0.f;
        if (k < 72) { int ci = k / 9, tap = k - ci * 9; v = d1w[(m * 8 + ci) * 9 + tap]; }
        wp[WP_D1 + i] = (_Float16)v;
    } else if (t < WP_TOTAL) {                       // d2: Kreal=Kpad=288
        int i = t - 8704; int m = i / 288, k = i % 288;
        int ci = k / 9, tap = k - ci * 9;
        wp[WP_D2 + i] = (_Float16)(d2w[(m * 32 + ci) * 9 + tap]);
    } else if (t < WP_TOTAL + 516) {
        stats[t - WP_TOTAL] = 0.f;
    }
}

// ---------------------------------------------------------------------------
// conv1 (1->16, 3x3, pad1, 256x256) fused with maxpool2 + exact GELU -> f16
// ---------------------------------------------------------------------------
__global__ __launch_bounds__(256)
void conv1_pool_gelu(const float* __restrict__ x, const float* __restrict__ w1,
                     const float* __restrict__ b1, _Float16* __restrict__ out1)
{
    __shared__ float sw[144];
    __shared__ float sb[16];
    int tid = threadIdx.x;
    if (tid < 144) sw[tid] = w1[tid];
    if (tid < 16)  sb[tid] = b1[tid];
    __syncthreads();

    int t  = blockIdx.x * 256 + tid;          // 32*128*128 threads
    int bb = t >> 14, rr = t & 16383, py = rr >> 7, px = rr & 127;

    float vals[4][4];
    #pragma unroll
    for (int r = 0; r < 4; ++r) {
        int iy = 2 * py - 1 + r;
        #pragma unroll
        for (int cix = 0; cix < 4; ++cix) {
            int ix = 2 * px - 1 + cix;
            vals[r][cix] = (iy >= 0 && iy < 256 && ix >= 0 && ix < 256)
                               ? x[bb * 65536 + iy * 256 + ix] : 0.f;
        }
    }
    #pragma unroll
    for (int c = 0; c < 16; ++c) {
        float m = -3.4e38f;
        #pragma unroll
        for (int sy = 0; sy < 2; ++sy)
            #pragma unroll
            for (int sx = 0; sx < 2; ++sx) {
                float s = sb[c];
                #pragma unroll
                for (int dy = 0; dy < 3; ++dy)
                    #pragma unroll
                    for (int dx = 0; dx < 3; ++dx)
                        s += vals[sy + dy][sx + dx] * sw[c * 9 + dy * 3 + dx];
                m = fmaxf(m, s);
            }
        out1[((bb * 16 + c) * 128 + py) * 128 + px] = (_Float16)gelu_exact(m);
    }
}

// ---------------------------------------------------------------------------
// Implicit-GEMM conv via v_wmma_f32_16x16x32_f16.
// WG stages an im2col panel [TOK][Kpad] + packed weights [Coutpad][Kpad] in
// LDS (weights via async global->LDS b128 when available). Each wave owns
// NSUB 16x16 C tiles; per k-step one A fragment feeds NSUB WMMAs.
// ---------------------------------------------------------------------------
template<int CIN, int COUT_R, int COUTP, int KS, int H, int W,
         bool UPS, bool ACT_GELU, bool F16OUT, int BLOCK, int NSUB>
__global__ __launch_bounds__(BLOCK)
void conv_wmma(const _Float16* __restrict__ in, const _Float16* __restrict__ wp,
               const float* __restrict__ bias, _Float16* __restrict__ out16,
               float* __restrict__ out32)
{
    constexpr int KREAL  = CIN * KS * KS;
    constexpr int KSTEPS = (KREAL + 31) / 32;
    constexpr int KPAD   = KSTEPS * 32;
    constexpr int KP     = KPAD + 8;          // row pad: bank de-stride, keeps 16B align
    constexpr int WAVES  = BLOCK / 32;
    constexpr int MTILES = COUTP / 16;
    constexpr int SUBW   = WAVES / MTILES;
    constexpr int TOK    = SUBW * NSUB * 16;
    constexpr int HIN    = UPS ? H / 2 : H;
    constexpr int WIN    = UPS ? W / 2 : W;
    constexpr int HW     = H * W;
    static_assert(WAVES % MTILES == 0, "");
    static_assert((TOK & (TOK - 1)) == 0, "TOK must be pow2");

    __shared__ alignas(16) _Float16 sW[COUTP * KP];
    __shared__ alignas(16) _Float16 sB[TOK * KP];

    const int tid     = threadIdx.x;
    const int tokBase = blockIdx.x * TOK;

    // stage packed weights: 16B chunks, async DMA to LDS when available
    constexpr int WCHUNK = KPAD / 8;
    for (int e = tid; e < COUTP * WCHUNK; e += BLOCK) {
        int m = e / WCHUNK, j = e - m * WCHUNK;
#if ASYNC_LDS
        __builtin_amdgcn_global_load_async_to_lds_b128(
            (gv4i*)(wp + m * KPAD + j * 8),
            (lv4i*)(&sW[m * KP + j * 8]), 0, 0);
#else
        *(v8h*)&sW[m * KP + j * 8] = *(const v8h*)(wp + m * KPAD + j * 8);
#endif
    }
    // stage im2col panel (k outer, token inner -> coalesced global reads)
    for (int e = tid; e < TOK * KPAD; e += BLOCK) {
        int k = e / TOK, tt = e & (TOK - 1);
        int token = tokBase + tt;
        int bb = token / HW, rr = token - bb * HW;
        int y = rr / W, x = rr - (rr / W) * W;
        _Float16 v = (_Float16)0.0f;
        if (k < KREAL) {
            int ci, iy, ix; bool ok = true;
            if (KS == 3) {
                ci = k / 9; int tap = k - ci * 9;
                int dy = tap / 3, dx = tap - dy * 3;
                iy = y + dy - 1; ix = x + dx - 1;
                ok = (iy >= 0) && (iy < H) && (ix >= 0) && (ix < W);
            } else { ci = k; iy = y; ix = x; }
            if (ok) {
                int ry = UPS ? (iy >> 1) : iy;
                int rx = UPS ? (ix >> 1) : ix;
                v = in[((bb * CIN + ci) * HIN + ry) * WIN + rx];
            }
        }
        sB[tt * KP + k] = v;
        // speculative prefetch of the next panel's base pixels (L2-resident)
        if (e < TOK) {
            int tok2 = tokBase + TOK + tt;
            if (tok2 < 32 * HW) {
                int b2 = tok2 / HW, r2 = tok2 - b2 * HW;
                int y2 = r2 / W, x2 = r2 - (r2 / W) * W;
                int ry2 = UPS ? (y2 >> 1) : y2;
                int rx2 = UPS ? (x2 >> 1) : x2;
                __builtin_prefetch(&in[((b2 * CIN) * HIN + ry2) * WIN + rx2], 0, 1);
            }
        }
    }
    wait_async_lds();
    __syncthreads();

    const int wave = tid >> 5, lane = tid & 31;
    const int mtile = wave / SUBW, sw_ = wave % SUBW;
    const int l15 = lane & 15, lh = lane >> 4;

    v8f c[NSUB];
    #pragma unroll
    for (int j = 0; j < NSUB; ++j) c[j] = (v8f){};

    #pragma unroll
    for (int ks = 0; ks < KSTEPS; ++ks) {
        // A 16x32 f16: lanes 0-15 hold K {0..7,16..23}, lanes 16-31 hold {8..15,24..31}
        const _Float16* wrow = &sW[(mtile * 16 + l15) * KP + ks * 32 + lh * 8];
        v8h alo = *(const v8h*)(wrow);
        v8h ahi = *(const v8h*)(wrow + 16);
        v16h a = __builtin_shufflevector(alo, ahi, 0,1,2,3,4,5,6,7,8,9,10,11,12,13,14,15);
        #pragma unroll
        for (int j = 0; j < NSUB; ++j) {
            // B 32x16 f16: lanes 0-15 hold K {0..7,16..23}, lanes 16-31 hold {8..15,24..31}
            const _Float16* brow =
                &sB[((sw_ * NSUB + j) * 16 + l15) * KP + ks * 32 + lh * 8];
            v8h blo = *(const v8h*)(brow);
            v8h bhi = *(const v8h*)(brow + 16);
            v16h b = __builtin_shufflevector(blo, bhi,
                                             0,1,2,3,4,5,6,7,8,9,10,11,12,13,14,15);
            c[j] = __builtin_amdgcn_wmma_f32_16x16x32_f16(false, a, false, b,
                                                          (short)0, c[j], false, false);
        }
    }

    // epilogue: C VGPR g -> (M = g + 8*lh, N = lane&15)
    #pragma unroll
    for (int j = 0; j < NSUB; ++j) {
        int token = tokBase + (sw_ * NSUB + j) * 16 + l15;
        int bb = token / HW, rr = token - bb * HW;
        int y = rr / W, x = rr - (rr / W) * W;
        #pragma unroll
        for (int g = 0; g < 8; ++g) {
            int cout = mtile * 16 + g + lh * 8;
            if (cout < COUT_R) {
                float v = c[j][g] + bias[cout];
                if (ACT_GELU) v = gelu_exact(v);
                int idx = ((bb * COUT_R + cout) * H + y) * W + x;
                if (F16OUT) out16[idx] = (_Float16)v;
                else        out32[idx] = v;
            }
        }
    }
}

// ---------------------------------------------------------------------------
// maxpool2 + GroupNorm(4 groups) over conv2 output; one WG per (batch, group).
// ---------------------------------------------------------------------------
__global__ __launch_bounds__(256)
void gn_pool_kernel(const _Float16* __restrict__ c2, _Float16* __restrict__ hn)
{
    __shared__ float rs_[256], rss[256];
    const int tid = threadIdx.x;
    const int bb = blockIdx.x >> 2, g = blockIdx.x & 3;
    const int N = 8 * 64 * 64;

    float s = 0.f, ss = 0.f;
    for (int i = tid; i < N; i += 256) {
        int lc = i >> 12, y = (i >> 6) & 63, x = i & 63;
        int cc = g * 8 + lc;
        int base = ((bb * 32 + cc) * 128 + 2 * y) * 128 + 2 * x;
        float v = fmaxf(fmaxf((float)c2[base], (float)c2[base + 1]),
                        fmaxf((float)c2[base + 128], (float)c2[base + 129]));
        s += v; ss += v * v;
    }
    rs_[tid] = s; rss[tid] = ss;
    __syncthreads();
    for (int st = 128; st > 0; st >>= 1) {
        if (tid < st) { rs_[tid] += rs_[tid + st]; rss[tid] += rss[tid + st]; }
        __syncthreads();
    }
    float mean = rs_[0] / (float)N;
    float var  = rss[0] / (float)N - mean * mean;
    float rstd = rsqrtf(var + 1e-5f);

    for (int i = tid; i < N; i += 256) {
        int lc = i >> 12, y = (i >> 6) & 63, x = i & 63;
        int cc = g * 8 + lc;
        int base = ((bb * 32 + cc) * 128 + 2 * y) * 128 + 2 * x;
        float v = fmaxf(fmaxf((float)c2[base], (float)c2[base + 1]),
                        fmaxf((float)c2[base + 128], (float)c2[base + 129]));
        hn[((bb * 32 + cc) * 64 + y) * 64 + x] = (_Float16)((v - mean) * rstd);
    }
}

// ---------------------------------------------------------------------------
// Residual LFQ: one wave per token, lanes cover the 256 codes (8 each).
// ---------------------------------------------------------------------------
__global__ __launch_bounds__(256)
void lfq_kernel(const float* __restrict__ h, _Float16* __restrict__ qb,
                float* __restrict__ dout, float* __restrict__ stats)
{
    __shared__ float sProb[512];
    __shared__ float sPC[4];          // pse[2], commit[2]
    const int tid = threadIdx.x;
    for (int j = tid; j < 512; j += 256) sProb[j] = 0.f;
    if (tid < 4) sPC[tid] = 0.f;
    __syncthreads();

    const int wave = tid >> 5, lane = tid & 31;
    const int token = blockIdx.x * 8 + wave;
    const int bb = token >> 12, rr = token & 4095, y = rr >> 6, x = rr & 63;

    float rv[8];
    #pragma unroll
    for (int i = 0; i < 8; ++i) rv[i] = h[((bb * 8 + i) * 64 + y) * 64 + x];
    float qsum[8] = {0.f, 0.f, 0.f, 0.f, 0.f, 0.f, 0.f, 0.f};

    #pragma unroll
    for (int q = 0; q < 2; ++q) {
        float qh[8]; int idx = 0; float S = 0.f;
        #pragma unroll
        for (int i = 0; i < 8; ++i) {
            qh[i] = rv[i] > 0.f ? 1.f : -1.f;
            if (rv[i] > 0.f) idx |= (1 << i);
            S += rv[i];
        }
        float loc[8];
        #pragma unroll
        for (int jj = 0; jj < 8; ++jj) {
            int code = lane + jj * 32;
            float sel = 0.f;
            #pragma unroll
            for (int i = 0; i < 8; ++i) if (code & (1 << i)) sel += rv[i];
            loc[jj] = 200.0f * (2.f * sel - S);
        }
        float mx = loc[0];
        #pragma unroll
        for (int jj = 1; jj < 8; ++jj) mx = fmaxf(mx, loc[jj]);
        #pragma unroll
        for (int off = 16; off > 0; off >>= 1) mx = fmaxf(mx, __shfl_xor(mx, off, 32));
        float es = 0.f;
        #pragma unroll
        for (int jj = 0; jj < 8; ++jj) es += expf(loc[jj] - mx);
        #pragma unroll
        for (int off = 16; off > 0; off >>= 1) es += __shfl_xor(es, off, 32);
        float logE = logf(es);
        float ent = 0.f;
        #pragma unroll
        for (int jj = 0; jj < 8; ++jj) {
            float lp = loc[jj] - mx - logE;
            float p  = expf(lp);
            ent -= p * lp;
            atomicAdd(&sProb[q * 256 + lane + jj * 32], p);
        }
        #pragma unroll
        for (int off = 16; off > 0; off >>= 1) ent += __shfl_xor(ent, off, 32);
        if (lane == 0) {
            atomicAdd(&sPC[q], ent);
            float cm = 0.f;
            #pragma unroll
            for (int i = 0; i < 8; ++i) { float d = rv[i] - qh[i]; cm += d * d; }
            atomicAdd(&sPC[2 + q], cm);
            dout[IDX_OFF + token * 2 + q] = (float)idx;
        }
        #pragma unroll
        for (int i = 0; i < 8; ++i) { qsum[i] += qh[i]; rv[i] -= qh[i]; }
    }
    if (lane == 0) {
        #pragma unroll
        for (int i = 0; i < 8; ++i)
            qb[((bb * 8 + i) * 64 + y) * 64 + x] = (_Float16)qsum[i];
    }
    __syncthreads();
    for (int j = tid; j < 512; j += 256) atomicAdd(&stats[4 + j], sProb[j]);
    if (tid < 4) atomicAdd(&stats[tid], sPC[tid]);
}

// ---------------------------------------------------------------------------
// Final aux scalar.
// ---------------------------------------------------------------------------
__global__ __launch_bounds__(256)
void aux_final(const float* __restrict__ stats, float* __restrict__ dout)
{
    __shared__ float red[256];
    const int tid = threadIdx.x;
    const float Ntok = (float)NTOK_LFQ;
    float cbe[2];
    for (int q = 0; q < 2; ++q) {
        __syncthreads();
        float ap = stats[4 + q * 256 + tid] / Ntok;
        red[tid] = -ap * logf(ap + 1e-10f);
        __syncthreads();
        for (int st = 128; st > 0; st >>= 1) {
            if (tid < st) red[tid] += red[tid + st];
            __syncthreads();
        }
        cbe[q] = red[0];
    }
    if (tid == 0) {
        float aux = 0.f;
        for (int q = 0; q < 2; ++q) {
            float pse    = stats[q] / Ntok;
            float commit = stats[2 + q] / (Ntok * 8.0f);
            aux += 0.1f * (pse - 1.0f * cbe[q]) + commit;
        }
        dout[AUX_POS] = aux;
    }
}

// ---------------------------------------------------------------------------
// d3 (16->1, 3x3, pad1) on upsampled 256x256 grid + clip -> d_out[0:Y_ELEMS]
// ---------------------------------------------------------------------------
__global__ __launch_bounds__(256)
void d3_clip_kernel(const _Float16* __restrict__ d2o, const float* __restrict__ w,
                    const float* __restrict__ bconst, float* __restrict__ yout)
{
    __shared__ float sw[144];
    __shared__ float sb;
    const int tid = threadIdx.x;
    if (tid < 144) sw[tid] = w[tid];
    if (tid == 0)  sb = bconst[0];
    __syncthreads();

    int t = blockIdx.x * 256 + tid;           // 32*256*256
    int bb = t >> 16, rr = t & 65535, y = rr >> 8, x = rr & 255;
    float acc = sb;
    #pragma unroll
    for (int ci = 0; ci < 16; ++ci) {
        #pragma unroll
        for (int dy = 0; dy < 3; ++dy) {
            int iy = y + dy - 1;
            if (iy < 0 || iy >= 256) continue;
            int ry = iy >> 1;
            #pragma unroll
            for (int dx = 0; dx < 3; ++dx) {
                int ix = x + dx - 1;
                if (ix < 0 || ix >= 256) continue;
                acc += (float)d2o[((bb * 16 + ci) * 128 + ry) * 128 + (ix >> 1)]
                       * sw[ci * 9 + dy * 3 + dx];
            }
        }
    }
    yout[t] = fminf(1.f, fmaxf(-1.f, acc));
}

// ---------------------------------------------------------------------------
extern "C" void kernel_launch(void* const* d_in, const int* in_sizes, int n_in,
                              void* d_out, int out_size, void* d_ws, size_t ws_size,
                              hipStream_t stream)
{
    const float* x   = (const float*)d_in[0];
    const float* w1  = (const float*)d_in[1];
    const float* b1  = (const float*)d_in[2];
    const float* w2  = (const float*)d_in[3];
    const float* b2  = (const float*)d_in[4];
    const float* w3  = (const float*)d_in[5];
    const float* b3  = (const float*)d_in[6];
    const float* d1w = (const float*)d_in[7];
    const float* db1 = (const float*)d_in[8];
    const float* d2w = (const float*)d_in[9];
    const float* db2 = (const float*)d_in[10];
    const float* d3w = (const float*)d_in[11];
    const float* db3 = (const float*)d_in[12];
    float* out = (float*)d_out;

    _Float16* wp    = (_Float16*)d_ws;
    float*    stats = (float*)((char*)d_ws + 32768);
    char*     bufA  = (char*)d_ws + 36864;
    _Float16* out1  = (_Float16*)bufA;                       // 32x16x128x128 f16
    _Float16* hn    = (_Float16*)bufA;                       // 32x32x64x64 f16 (reuse)
    float*    h32   = (float*)   (bufA + 8388608);           // 32x8x64x64 f32
    _Float16* qb    = (_Float16*)(bufA + 12582912);          // 32x8x64x64 f16
    char*     bufB  = bufA + 16777216;
    _Float16* c2o   = (_Float16*)bufB;                       // 32x32x128x128 f16
    _Float16* d1o   = (_Float16*)bufB;                       // 32x32x64x64 f16 (reuse)
    _Float16* d2o   = (_Float16*)(bufB + 8388608);           // 32x16x128x128 f16

    // 1) pack weights to f16 im2col layout + zero stat accumulators
    prep_pack<<<55, 256, 0, stream>>>(w2, w3, d1w, d2w, wp, stats);

    // 2) conv1 + maxpool + gelu  (VALU; K=9 too skinny for WMMA tiles)
    conv1_pool_gelu<<<2048, 256, 0, stream>>>(x, w1, b1, out1);

    // 3) conv2 16->32, 3x3 @128x128  (WMMA, K=144->160, 5 k-steps, 2 C-tiles/wave)
    conv_wmma<16, 32, 32, 3, 128, 128, false, false, true, 256, 2>
        <<<4096, 256, 0, stream>>>(out1, wp + WP_C2, b2, c2o, nullptr);

    // 4) maxpool + groupnorm
    gn_pool_kernel<<<128, 256, 0, stream>>>(c2o, hn);

    // 5) conv3 1x1 32->8 (WMMA, single K=32 step, f32 out for LFQ)
    conv_wmma<32, 8, 16, 1, 64, 64, false, false, false, 128, 2>
        <<<1024, 128, 0, stream>>>(hn, wp + WP_C3, b3, nullptr, h32);

    // 6) residual LFQ: indices + aux stats + quantized f16
    lfq_kernel<<<16384, 256, 0, stream>>>(h32, qb, out, stats);
    aux_final<<<1, 256, 0, stream>>>(stats, out);

    // 7) d1 8->32, 3x3 @64x64 (WMMA, K=72->96, 3 k-steps, 2 C-tiles/wave)
    conv_wmma<8, 32, 32, 3, 64, 64, false, false, true, 256, 2>
        <<<1024, 256, 0, stream>>>(qb, wp + WP_D1, db1, d1o, nullptr);

    // 8) d2 32->16, 3x3 on upsampled 128x128 grid + gelu (WMMA, K=288, 9 k-steps)
    conv_wmma<32, 16, 16, 3, 128, 128, true, true, true, 128, 1>
        <<<8192, 128, 0, stream>>>(d1o, wp + WP_D2, db2, d2o, nullptr);

    // 9) d3 16->1 on upsampled 256x256 grid + clip -> final y
    d3_clip_kernel<<<8192, 256, 0, stream>>>(d2o, d3w, db3, out);
}
